// KernelLinear_60292750901529
// MI455X (gfx1250) — compile-verified, matrix operand order checked by reading, not run on hardware
//
#include <hip/hip_runtime.h>
#include <hip/hip_bf16.h>

// ---------------------------------------------------------------------------
// out[b,o] = log(exp(-sqrt(max(x2[b] + w2[o] - 2*x.w, 0)) / 2))
//          = -0.5 * sqrt(max(d2, 0))     (exact to <=2 ulp: the argument of
//            exp is in [-13, 0], far from underflow, so log(exp(z)) == z up
//            to float rounding; dropping the pair cuts TRANS-pipe work 3x).
// sqrt uses raw v_sqrt_f32 (__builtin_amdgcn_sqrtf): argument is max(d2,0),
// always a normal float here, so the IEEE denorm-guard expansion that
// __builtin_sqrtf emits (~6 VALU/sqrt) is dead weight.
// B=65536, K=128, O=1024.  bf16 WMMA GEMM for the cross term (f32 accum),
// exact f32 norms as correction terms.  ~288 MB HBM traffic -> ~12.4 us.
// ---------------------------------------------------------------------------

#define BATCH 65536
#define KDIM  128
#define OUTF  1024
#define TILE  128
// LDS row stride in uints (136 bf16 = 68 uints; +8 bf16 pad vs 128 to break
// the 64-bank alignment when 16 lanes read 16 different rows at same K).
#define LDU   68

typedef __attribute__((ext_vector_type(16))) __bf16        v16bf;
typedef __attribute__((ext_vector_type(8)))  float         v8f;
typedef __attribute__((ext_vector_type(8)))  unsigned int  v8u;

static __device__ __forceinline__ unsigned int pack_bf16(float a, float b) {
    // round-to-nearest-even f32 -> bf16, packed pair
    unsigned int ua = __float_as_uint(a);
    unsigned int ub = __float_as_uint(b);
    ua = (ua + 0x7fffu + ((ua >> 16) & 1u)) >> 16;
    ub = (ub + 0x7fffu + ((ub >> 16) & 1u)) >> 16;
    return ua | (ub << 16);
}

// Gather one 16x32 bf16 fragment chunk pair from LDS (two ds_load_b128).
static __device__ __forceinline__ v16bf load_frag(const unsigned int* p) {
    uint4 a = *(const uint4*)(p);
    uint4 b = *(const uint4*)(p + 8);
    v8u u = { a.x, a.y, a.z, a.w, b.x, b.y, b.z, b.w };
    return __builtin_bit_cast(v16bf, u);
}

// ---------------------------------------------------------------------------
// Kernel 1: squared row norms of X (65536 rows) and W (1024 rows).
// One wave32 per row: each lane loads a float4 (coalesced 512B/wave),
// then a 5-step shfl_xor reduction.
// ---------------------------------------------------------------------------
__global__ void row_norms_kernel(const float* __restrict__ x,
                                 const float* __restrict__ w,
                                 float* __restrict__ x2,
                                 float* __restrict__ w2) {
    const int gwave = (blockIdx.x * blockDim.x + threadIdx.x) >> 5;
    const int lane  = threadIdx.x & 31;
    if (gwave >= BATCH + OUTF) return;

    const float* __restrict__ src;
    float* __restrict__ dst;
    if (gwave < BATCH) { src = x + (size_t)gwave * KDIM;           dst = x2 + gwave; }
    else               { src = w + (size_t)(gwave - BATCH) * KDIM; dst = w2 + (gwave - BATCH); }

    float4 v = ((const float4*)src)[lane];               // 128 floats per row
    float  s = v.x * v.x + v.y * v.y + v.z * v.z + v.w * v.w;
#pragma unroll
    for (int off = 16; off > 0; off >>= 1)
        s += __shfl_xor(s, off, 32);
    if (lane == 0) *dst = s;
}

// ---------------------------------------------------------------------------
// Kernel 2: tiled GEMM + distance epilogue.
// Block = 256 threads (8 waves), tile = 128(M) x 128(N), full K=128 in LDS.
// Wave (wm,wn): wm in 0..3 -> 32 M rows, wn in 0..1 -> 64 N cols.
// 4 K-steps x (2x4) subtiles = 32 v_wmma_f32_16x16x32_bf16 per wave.
// ---------------------------------------------------------------------------
__global__ void __launch_bounds__(256)
rbf_gemm_kernel(const float* __restrict__ x,
                const float* __restrict__ w,
                const float* __restrict__ x2g,
                const float* __restrict__ w2g,
                float* __restrict__ out) {
    __shared__ unsigned int sA[TILE * LDU];   // 34816 B, bf16 X tile
    __shared__ unsigned int sB[TILE * LDU];   // 34816 B, bf16 W tile
    __shared__ float sX2[TILE];
    __shared__ float sW2[TILE];

    const int tid = threadIdx.x;
    const int m0  = blockIdx.y * TILE;        // batch rows
    const int n0  = blockIdx.x * TILE;        // output features

    // ---- stage tiles: f32 global -> bf16 LDS (coalesced float4 loads) ----
#pragma unroll
    for (int i = 0; i < 16; ++i) {
        const int idx = tid + i * 256;        // 0..4095 over 128 rows x 32 float4
        const int row = idx >> 5;
        const int c4  = idx & 31;
        float4 va = ((const float4*)(x + (size_t)(m0 + row) * KDIM))[c4];
        sA[row * LDU + c4 * 2]     = pack_bf16(va.x, va.y);
        sA[row * LDU + c4 * 2 + 1] = pack_bf16(va.z, va.w);
        float4 vb = ((const float4*)(w + (size_t)(n0 + row) * KDIM))[c4];
        sB[row * LDU + c4 * 2]     = pack_bf16(vb.x, vb.y);
        sB[row * LDU + c4 * 2 + 1] = pack_bf16(vb.z, vb.w);
    }
    if (tid < TILE)       sX2[tid]        = x2g[m0 + tid];
    else                  sW2[tid - TILE] = w2g[n0 + tid - TILE];
    __syncthreads();

    // ---- wave-level WMMA compute ----
    const int lane = tid & 31;
    const int wave = tid >> 5;
    const int wm   = wave & 3;                // M slot: rows wm*32 .. +31
    const int wn   = wave >> 2;               // N slot: cols wn*64 .. +63
    const int lm   = lane & 15;
    const int kh   = lane >> 4;               // half-wave K-chunk selector

    v8f acc[2][4];
#pragma unroll
    for (int s = 0; s < 2; ++s)
#pragma unroll
        for (int t = 0; t < 4; ++t)
            acc[s][t] = (v8f){0.f, 0.f, 0.f, 0.f, 0.f, 0.f, 0.f, 0.f};

#pragma unroll
    for (int ks = 0; ks < 4; ++ks) {          // K = ks*32 .. +31
        v16bf afrag[2];
#pragma unroll
        for (int s = 0; s < 2; ++s) {
            const int row = wm * 32 + s * 16 + lm;
            afrag[s] = load_frag(&sA[row * LDU + ks * 16 + kh * 4]);
        }
        v16bf bfrag[4];
#pragma unroll
        for (int t = 0; t < 4; ++t) {
            const int row = wn * 64 + t * 16 + lm;
            bfrag[t] = load_frag(&sB[row * LDU + ks * 16 + kh * 4]);
        }
#pragma unroll
        for (int s = 0; s < 2; ++s)
#pragma unroll
            for (int t = 0; t < 4; ++t)
                acc[s][t] = __builtin_amdgcn_wmma_f32_16x16x32_bf16(
                    false, afrag[s], false, bfrag[t],
                    (short)0, acc[s][t], false, false);
    }

    // ---- epilogue: d2 = x2 + w2 - 2*dot ; out = -0.5*sqrt(max(d2,0)) ----
    // (== log(exp(-sqrt(d2)/2)) to within float rounding; see header note)
    // C/D layout: VGPR r -> M = r + 8*(lane>>4), N = lane&15.
#pragma unroll
    for (int s = 0; s < 2; ++s) {
#pragma unroll
        for (int t = 0; t < 4; ++t) {
            const int n_l = wn * 64 + t * 16 + lm;
            const float w2v = sW2[n_l];
#pragma unroll
            for (int r = 0; r < 8; ++r) {
                const int m_l = wm * 32 + s * 16 + r + 8 * kh;
                float d2   = sX2[m_l] + w2v - 2.0f * acc[s][t][r];
                float dist = __builtin_amdgcn_sqrtf(fmaxf(d2, 0.0f));
                out[(size_t)(m0 + m_l) * OUTF + (n0 + n_l)] = -0.5f * dist;
            }
        }
    }
}

// ---------------------------------------------------------------------------
extern "C" void kernel_launch(void* const* d_in, const int* in_sizes, int n_in,
                              void* d_out, int out_size, void* d_ws, size_t ws_size,
                              hipStream_t stream) {
    const float* x = (const float*)d_in[0];   // [65536, 128] f32
    const float* w = (const float*)d_in[1];   // [1024, 128]  f32
    float* out = (float*)d_out;               // [65536, 1024] f32
    float* x2  = (float*)d_ws;                // 65536 f32
    float* w2  = x2 + BATCH;                  // 1024 f32  (ws use: ~266 KB)

    // 66560 waves, 32 threads each -> 8320 blocks of 256.
    const int nrows = BATCH + OUTF;
    const int nblk  = (nrows * 32 + 255) / 256;
    hipLaunchKernelGGL(row_norms_kernel, dim3(nblk), dim3(256), 0, stream,
                       x, w, x2, w2);

    // Grid: x = N tiles (8), y = M tiles (512); adjacent blocks share X tile
    // in L2 (entire X = 32 MB fits in 192 MB L2).
    hipLaunchKernelGGL(rbf_gemm_kernel, dim3(OUTF / TILE, BATCH / TILE),
                       dim3(256), 0, stream, x, w, x2, w2, out);
}
